// LSTM_65781719105691
// MI455X (gfx1250) — matrix operand order repaired and checked by reading, not compiled, not run on hardware
//
#include <hip/hip_runtime.h>
#include <hip/hip_bf16.h>

typedef __attribute__((ext_vector_type(16))) _Float16 v16h;
typedef __attribute__((ext_vector_type(8)))  _Float16 v8h;
typedef __attribute__((ext_vector_type(8)))  float    v8f;

#define Hsz 512
#define Tsz 256
#define Bsz 256
#define BM  16          // batch rows per workgroup
#define NCH0 17         // K chunks layer0: 512/32 + 1 (x|1 ext)
#define NCH1 33         // K chunks layer1: 16 (h0) + 16 (h1) + 1 (bias ext)
#define HSTR0 544       // LDS row stride (halfs) layer0 (512 + 32 ext)

#define LOG2E 1.4426950408889634f

// Branch-free nonlinearities on hardware TRANS ops (v_exp_f32 / v_rcp_f32).
__device__ __forceinline__ float sigf(float x) {
  return __builtin_amdgcn_rcpf(1.0f + __builtin_amdgcn_exp2f(-LOG2E * x));
}
__device__ __forceinline__ float tanh_fast(float x) {
  // tanh(x) = 2*sigmoid(2x) - 1
  return fmaf(2.0f, __builtin_amdgcn_rcpf(1.0f + __builtin_amdgcn_exp2f(-2.0f * LOG2E * x)), -1.0f);
}

__device__ __forceinline__ v8f wmma32(v16h a, v16h b, v8f c) {
  // D = A(16x32 f16) * B(32x16 f16) + C(16x16 f32)
  return __builtin_amdgcn_wmma_f32_16x16x32_f16(false, a, false, b, (short)0, c, false, false);
}

// A operand: row `rowbase` (one batch row), K-chunk kc.
// lanes 0-15 hold K = kc*32 + {0..7, 16..23}; lanes 16-31 hold K = kc*32 + {8..15, 24..31}.
__device__ __forceinline__ v16h load_a16(const _Float16* rowbase, int kc, int hi) {
  v8h lo = *(const v8h*)(rowbase + kc * 32 + hi * 8);
  v8h hv = *(const v8h*)(rowbase + kc * 32 + 16 + hi * 8);
  return __builtin_shufflevector(lo, hv, 0, 1, 2, 3, 4, 5, 6, 7, 8, 9, 10, 11, 12, 13, 14, 15);
}

// ---------------------------------------------------------------------------
// Pack logical B matrix [K x 2048] into per-lane WMMA layout:
// out[((n*nch + kc)*32 + lane)*16 + i] = W[kc*32 + (i>>3)*16 + (lane>>4)*8 + (i&7)][col = n*16 + (lane&15)]
// mode 0 (layer0): W[k][col] = k<512 ? w_hh0[col,k] : (k<517 ? w_ih0[col,k-512] : (k==517 ? b0[col] : 0))
// mode 1 (layer1): W[k][col] = k<512 ? w_ih1[col,k] : (k<1024 ? w_hh1[col,k-512] : (k==1024 ? b1[col] : 0))
// ---------------------------------------------------------------------------
__global__ void pack_b_kernel(const float* __restrict__ A1, const float* __restrict__ A2,
                              const float* __restrict__ bias, _Float16* __restrict__ out,
                              int nch, int mode, int total) {
  int i = blockIdx.x * blockDim.x + threadIdx.x;
  if (i >= total) return;
  int hIdx = i & 15;
  int lane = (i >> 4) & 31;
  int rem  = i >> 9;              // n*nch + kc
  int kc   = rem % nch;
  int n    = rem / nch;
  int col  = n * 16 + (lane & 15);
  int k    = kc * 32 + (hIdx >> 3) * 16 + (lane >> 4) * 8 + (hIdx & 7);
  float v = 0.0f;
  if (mode == 0) {
    if (k < 512)        v = A1[(size_t)col * 512 + k];
    else if (k < 517)   v = A2[col * 5 + (k - 512)];
    else if (k == 517)  v = bias[col];
  } else {
    if (k < 512)        v = A1[(size_t)col * 512 + k];
    else if (k < 1024)  v = A2[(size_t)col * 512 + (k - 512)];
    else if (k == 1024) v = bias[col];
  }
  out[i] = (_Float16)v;
}

// ---------------------------------------------------------------------------
// Layer 0: gates = [h_{t-1} | x_t,1] @ packedB0 ; LSTM cell; h0 -> LDS + global fp16
// grid = 16 (batch tiles), block = 512 (16 waves; wave w owns h cols [32w,32w+32))
// ---------------------------------------------------------------------------
__global__ __launch_bounds__(512, 1)
void lstm_layer0_kernel(const float* __restrict__ x_time,
                        const _Float16* __restrict__ pB0,
                        _Float16* __restrict__ h0g) {
  __shared__ _Float16 h_lds[BM * HSTR0];
  const int tid  = threadIdx.x;
  const int lane = tid & 31;
  const int w    = tid >> 5;
  const int mrow = lane & 15;
  const int hi   = lane >> 4;
  const int bt   = blockIdx.x;
  const int b0   = bt * BM;

  for (int i = tid; i < BM * HSTR0; i += 512) h_lds[i] = (_Float16)0.0f;

  float c_state[2][8];
#pragma unroll
  for (int j = 0; j < 2; ++j)
#pragma unroll
    for (int r = 0; r < 8; ++r) c_state[j][r] = 0.0f;

  const _Float16* arow = &h_lds[mrow * HSTR0];

  for (int t = 0; t < Tsz; ++t) {
    // stage [x_t | 1 | 0...] into ext K-chunk (cols 512..543)
    for (int i = tid; i < BM * 32; i += 512) {
      int m = i >> 5, kk = i & 31;
      float v = 0.0f;
      if (kk < 5)       v = x_time[((size_t)(b0 + m) * Tsz + t) * 5 + kk];
      else if (kk == 5) v = 1.0f;
      h_lds[m * HSTR0 + 512 + kk] = (_Float16)v;
    }
    __syncthreads();

    float nh[2][8];
#pragma unroll
    for (int j = 0; j < 2; ++j) {
      int nb = 2 * w + j;   // i-gate tile index; f/g/o at +32/+64/+96
      const _Float16* pBi = pB0 + (size_t)(nb)      * NCH0 * 512 + lane * 16;
      const _Float16* pBf = pB0 + (size_t)(32 + nb) * NCH0 * 512 + lane * 16;
      const _Float16* pBg = pB0 + (size_t)(64 + nb) * NCH0 * 512 + lane * 16;
      const _Float16* pBo = pB0 + (size_t)(96 + nb) * NCH0 * 512 + lane * 16;
      v8f ai = {}, af = {}, ag = {}, ao = {};
      // NOTE: unroll 1 — full unroll spills B operands to scratch (reg pressure).
#pragma unroll 1
      for (int kc = 0; kc < NCH0; ++kc) {
        v16h A  = load_a16(arow, kc, hi);
        v16h Bi = *(const v16h*)(pBi + kc * 512);
        v16h Bf = *(const v16h*)(pBf + kc * 512);
        v16h Bg = *(const v16h*)(pBg + kc * 512);
        v16h Bo = *(const v16h*)(pBo + kc * 512);
        ai = wmma32(A, Bi, ai);
        af = wmma32(A, Bf, af);
        ag = wmma32(A, Bg, ag);
        ao = wmma32(A, Bo, ao);
      }
#pragma unroll
      for (int r = 0; r < 8; ++r) {
        float iv = sigf(ai[r]);
        float fv = sigf(af[r]);
        float gv = tanh_fast(ag[r]);
        float ov = sigf(ao[r]);
        float cv = fv * c_state[j][r] + iv * gv;
        c_state[j][r] = cv;
        nh[j][r] = ov * tanh_fast(cv);
      }
    }
    __syncthreads();   // all waves done reading h_lds for this step
#pragma unroll
    for (int j = 0; j < 2; ++j) {
      int col = 32 * w + 16 * j + mrow;
#pragma unroll
      for (int r = 0; r < 8; ++r) {
        int m = r + 8 * hi;
        _Float16 hv = (_Float16)nh[j][r];
        h_lds[m * HSTR0 + col] = hv;
        h0g[(((size_t)bt * Tsz + t) * BM + m) * Hsz + col] = hv;
      }
    }
  }
}

// ---------------------------------------------------------------------------
// Layer 1 (fused input-proj + recurrence + fc1 reduction):
// gates = [h0_t | h1_{t-1} | 1] @ packedB1;  fc1 partials accumulated per step.
// ---------------------------------------------------------------------------
__global__ __launch_bounds__(512, 1)
void lstm_layer1_kernel(const _Float16* __restrict__ h0g,
                        const _Float16* __restrict__ pB1,
                        const float* __restrict__ fc1_w,
                        float* __restrict__ tp) {
  __shared__ _Float16 h_lds[BM * Hsz];
  __shared__ float tp_lds[BM][2];
  const int tid  = threadIdx.x;
  const int lane = tid & 31;
  const int w    = tid >> 5;
  const int mrow = lane & 15;
  const int hi   = lane >> 4;
  const int bt   = blockIdx.x;
  const int b0   = bt * BM;

  for (int i = tid; i < BM * Hsz; i += 512) h_lds[i] = (_Float16)0.0f;
  if (tid < 32) tp_lds[tid >> 1][tid & 1] = 0.0f;

  float c_state[2][8];
  float tpa[8][2];
#pragma unroll
  for (int r = 0; r < 8; ++r) { tpa[r][0] = 0.0f; tpa[r][1] = 0.0f; }
#pragma unroll
  for (int j = 0; j < 2; ++j)
#pragma unroll
    for (int r = 0; r < 8; ++r) c_state[j][r] = 0.0f;

  v16h abias = {};
  if (lane < 16) abias[0] = (_Float16)1.0f;   // K-row 0 of the bias chunk

  const _Float16* hrow = &h_lds[mrow * Hsz];
  __syncthreads();

  for (int t = 0; t < Tsz; ++t) {
    const _Float16* a0row = h0g + (((size_t)bt * Tsz + t) * BM + mrow) * Hsz;
    float nh[2][8];
#pragma unroll
    for (int j = 0; j < 2; ++j) {
      int nb = 2 * w + j;
      const _Float16* pBi = pB1 + (size_t)(nb)      * NCH1 * 512 + lane * 16;
      const _Float16* pBf = pB1 + (size_t)(32 + nb) * NCH1 * 512 + lane * 16;
      const _Float16* pBg = pB1 + (size_t)(64 + nb) * NCH1 * 512 + lane * 16;
      const _Float16* pBo = pB1 + (size_t)(96 + nb) * NCH1 * 512 + lane * 16;
      v8f ai = {}, af = {}, ag = {}, ao = {};
      // K chunks 0..15: A from h0 (global, fp16, L2-resident)
#pragma unroll 1
      for (int kc = 0; kc < 16; ++kc) {
        v16h A  = load_a16(a0row, kc, hi);
        v16h Bi = *(const v16h*)(pBi + kc * 512);
        v16h Bf = *(const v16h*)(pBf + kc * 512);
        v16h Bg = *(const v16h*)(pBg + kc * 512);
        v16h Bo = *(const v16h*)(pBo + kc * 512);
        ai = wmma32(A, Bi, ai);
        af = wmma32(A, Bf, af);
        ag = wmma32(A, Bg, ag);
        ao = wmma32(A, Bo, ao);
      }
      // K chunks 16..31: A from h1_{t-1} (LDS)
#pragma unroll 1
      for (int kc = 0; kc < 16; ++kc) {
        v16h A  = load_a16(hrow, kc, hi);
        v16h Bi = *(const v16h*)(pBi + (16 + kc) * 512);
        v16h Bf = *(const v16h*)(pBf + (16 + kc) * 512);
        v16h Bg = *(const v16h*)(pBg + (16 + kc) * 512);
        v16h Bo = *(const v16h*)(pBo + (16 + kc) * 512);
        ai = wmma32(A, Bi, ai);
        af = wmma32(A, Bf, af);
        ag = wmma32(A, Bg, ag);
        ao = wmma32(A, Bo, ao);
      }
      // bias chunk
      ai = wmma32(abias, *(const v16h*)(pBi + 32 * 512), ai);
      af = wmma32(abias, *(const v16h*)(pBf + 32 * 512), af);
      ag = wmma32(abias, *(const v16h*)(pBg + 32 * 512), ag);
      ao = wmma32(abias, *(const v16h*)(pBo + 32 * 512), ao);

      int col = 32 * w + 16 * j + mrow;
      float w10 = fc1_w[(size_t)t * Hsz + col];
      float w11 = fc1_w[(size_t)Tsz * Hsz + (size_t)t * Hsz + col];
#pragma unroll
      for (int r = 0; r < 8; ++r) {
        float iv = sigf(ai[r]);
        float fv = sigf(af[r]);
        float gv = tanh_fast(ag[r]);
        float ov = sigf(ao[r]);
        float cv = fv * c_state[j][r] + iv * gv;
        c_state[j][r] = cv;
        float hvf = ov * tanh_fast(cv);
        nh[j][r] = hvf;
        tpa[r][0] += hvf * w10;      // fused fc1 (time_preds) partial
        tpa[r][1] += hvf * w11;
      }
    }
    __syncthreads();
#pragma unroll
    for (int j = 0; j < 2; ++j) {
      int col = 32 * w + 16 * j + mrow;
#pragma unroll
      for (int r = 0; r < 8; ++r)
        h_lds[(r + 8 * hi) * Hsz + col] = (_Float16)nh[j][r];
    }
    __syncthreads();
  }

  // reduce fc1 partials across lanes/waves (rows are disjoint per workgroup)
#pragma unroll
  for (int r = 0; r < 8; ++r) {
    int m = r + 8 * hi;
    atomicAdd(&tp_lds[m][0], tpa[r][0]);
    atomicAdd(&tp_lds[m][1], tpa[r][1]);
  }
  __syncthreads();
  if (tid < 32) tp[(b0 + (tid >> 1)) * 2 + (tid & 1)] = tp_lds[tid >> 1][tid & 1];
}

// ---------------------------------------------------------------------------
// Heads: stat_preds (fc2), add fc1 bias, concat, fc3.
// ---------------------------------------------------------------------------
__global__ void head_kernel(const float* __restrict__ x_stat, const float* __restrict__ tp,
                            const float* __restrict__ fc1_b, const float* __restrict__ fc2_w,
                            const float* __restrict__ fc2_b, const float* __restrict__ fc3_w,
                            const float* __restrict__ fc3_b, float* __restrict__ out) {
  int b = blockIdx.x * blockDim.x + threadIdx.x;
  if (b >= Bsz) return;
  float s0 = fc2_b[0], s1 = fc2_b[1];
#pragma unroll
  for (int d = 0; d < 3; ++d) {
    float xv = x_stat[b * 3 + d];
    s0 += xv * fc2_w[d];
    s1 += xv * fc2_w[3 + d];
  }
  float p[4] = { s0, s1, tp[b * 2] + fc1_b[0], tp[b * 2 + 1] + fc1_b[1] };
#pragma unroll
  for (int k = 0; k < 2; ++k) {
    float acc = fc3_b[k];
#pragma unroll
    for (int j4 = 0; j4 < 4; ++j4) acc += p[j4] * fc3_w[k * 4 + j4];
    out[b * 2 + k] = acc;
  }
}

extern "C" void kernel_launch(void* const* d_in, const int* in_sizes, int n_in,
                              void* d_out, int out_size, void* d_ws, size_t ws_size,
                              hipStream_t stream) {
  const float* x_time = (const float*)d_in[0];
  const float* x_stat = (const float*)d_in[1];
  const float* w_ih0  = (const float*)d_in[2];
  const float* w_hh0  = (const float*)d_in[3];
  const float* b0     = (const float*)d_in[4];
  const float* w_ih1  = (const float*)d_in[5];
  const float* w_hh1  = (const float*)d_in[6];
  const float* b1     = (const float*)d_in[7];
  const float* fc1_w  = (const float*)d_in[8];
  const float* fc1_b  = (const float*)d_in[9];
  const float* fc2_w  = (const float*)d_in[10];
  const float* fc2_b  = (const float*)d_in[11];
  const float* fc3_w  = (const float*)d_in[12];
  const float* fc3_b  = (const float*)d_in[13];
  float* out = (float*)d_out;

  // workspace layout
  const int nPB0 = 128 * NCH0 * 512;           // 1,114,112 halfs
  const int nPB1 = 128 * NCH1 * 512;           // 2,162,688 halfs
  char* ws = (char*)d_ws;
  _Float16* pB0 = (_Float16*)ws;
  _Float16* pB1 = (_Float16*)(ws + (size_t)nPB0 * 2);
  _Float16* h0g = (_Float16*)(ws + (size_t)nPB0 * 2 + (size_t)nPB1 * 2);
  float*    tp  = (float*)(ws + (size_t)nPB0 * 2 + (size_t)nPB1 * 2 +
                           (size_t)Bsz * Tsz * Hsz * 2);

  pack_b_kernel<<<(nPB0 + 255) / 256, 256, 0, stream>>>(w_hh0, w_ih0, b0, pB0, NCH0, 0, nPB0);
  pack_b_kernel<<<(nPB1 + 255) / 256, 256, 0, stream>>>(w_ih1, w_hh1, b1, pB1, NCH1, 1, nPB1);
  lstm_layer0_kernel<<<Bsz / BM, 512, 0, stream>>>(x_time, pB0, h0g);
  lstm_layer1_kernel<<<Bsz / BM, 512, 0, stream>>>(h0g, pB1, fc1_w, tp);
  head_kernel<<<1, 256, 0, stream>>>(x_stat, tp, fc1_b, fc2_w, fc2_b, fc3_w, fc3_b, out);
}
